// SelectiveSSM_12266426597758
// MI455X (gfx1250) — matrix operand order repaired and checked by reading, not compiled, not run on hardware
//
#include <hip/hip_runtime.h>
#include <hip/hip_bf16.h>
#include <math.h>

#define D_MODEL 1024
#define D_STATE 16
#define D_CONV  4
#define D_INNER 2048
#define BATCH   2
#define SEQ     2048
#define NTOK    (BATCH * SEQ)   // 4096 tokens
#define NCHUNK  16
#define CHUNK   (SEQ / NCHUNK)  // 128
#define NPROJ   (2 * D_STATE + 1) // 33

typedef _Float16 v16h __attribute__((ext_vector_type(16)));
typedef _Float16 v8h  __attribute__((ext_vector_type(8)));
typedef float    v8f  __attribute__((ext_vector_type(8)));

// ---------------------------------------------------------------------------
// 1. LayerNorm -> f16 activations (one block per token)
// ---------------------------------------------------------------------------
__global__ __launch_bounds__(256) void ln_to_f16_k(
    const float* __restrict__ x, const float* __restrict__ g,
    const float* __restrict__ b, _Float16* __restrict__ xn)
{
    __shared__ float red[256];
    const int t = blockIdx.x;
    const float* row = x + (size_t)t * D_MODEL;

    float s = 0.f;
    for (int i = threadIdx.x; i < D_MODEL; i += 256) s += row[i];
    red[threadIdx.x] = s; __syncthreads();
    for (int o = 128; o > 0; o >>= 1) {
        if (threadIdx.x < o) red[threadIdx.x] += red[threadIdx.x + o];
        __syncthreads();
    }
    const float mu = red[0] * (1.f / D_MODEL);
    __syncthreads();

    float v = 0.f;
    for (int i = threadIdx.x; i < D_MODEL; i += 256) {
        float d = row[i] - mu; v += d * d;
    }
    red[threadIdx.x] = v; __syncthreads();
    for (int o = 128; o > 0; o >>= 1) {
        if (threadIdx.x < o) red[threadIdx.x] += red[threadIdx.x + o];
        __syncthreads();
    }
    const float rstd = rsqrtf(red[0] * (1.f / D_MODEL) + 1e-5f);

    for (int i = threadIdx.x; i < D_MODEL; i += 256)
        xn[(size_t)t * D_MODEL + i] = (_Float16)((row[i] - mu) * rstd * g[i] + b[i]);
}

// ---------------------------------------------------------------------------
// 2. Weight transpose + f32->f16 convert: (K,N) row-major -> (N,K) row-major.
// ---------------------------------------------------------------------------
__global__ __launch_bounds__(256) void transpose_to_f16_k(
    const float* __restrict__ in, _Float16* __restrict__ out, int K, int N)
{
    size_t idx = (size_t)blockIdx.x * blockDim.x + threadIdx.x;
    if (idx >= (size_t)K * N) return;
    int k = (int)(idx / (size_t)N);
    int n = (int)(idx - (size_t)k * N);
    out[(size_t)n * K + k] = (_Float16)in[idx];
}

// ---------------------------------------------------------------------------
// 3. WMMA GEMM: C(MxN,f32) = A(MxK,f16 row-major) * Bt(NxK,f16 row-major)^T
//    [+ residual].  8 waves/block (4 in M x 2 in N), each wave -> 32x32 tile
//    (2 A-frags, 2 B-frags, 4 WMMAs per k-step => ~1KB fragment traffic/WMMA).
// ---------------------------------------------------------------------------
__global__ __launch_bounds__(256) void wmma_gemm_f16f32_k(
    const _Float16* __restrict__ A, const _Float16* __restrict__ Bt,
    float* __restrict__ C, const float* __restrict__ resid,
    int M, int N, int K)
{
    const int lane  = threadIdx.x & 31;
    const int wave  = threadIdx.x >> 5;
    const int hl    = lane >> 4;    // lane half: 0 or 1
    const int lr    = lane & 15;
    const int waveM = wave & 3;
    const int waveN = wave >> 2;

    const int m0 = blockIdx.y * 128 + waveM * 32;
    const int n0 = blockIdx.x * 64 + waveN * 32;
    if (m0 >= M || n0 >= N) return;

    const _Float16* pa0 = A  + (size_t)(m0 + lr) * K + hl * 8;
    const _Float16* pa1 = A  + (size_t)(m0 + 16 + lr) * K + hl * 8;
    const _Float16* pb0 = Bt + (size_t)(n0 + lr) * K + hl * 8;
    const _Float16* pb1 = Bt + (size_t)(n0 + 16 + lr) * K + hl * 8;

    v8f acc00 = {}, acc01 = {}, acc10 = {}, acc11 = {};

    for (int k0 = 0; k0 < K; k0 += 32) {
        v8h a0l = *(const v8h*)(pa0);
        v8h a0h = *(const v8h*)(pa0 + 16);
        v8h a1l = *(const v8h*)(pa1);
        v8h a1h = *(const v8h*)(pa1 + 16);
        v8h b0l = *(const v8h*)(pb0);
        v8h b0h = *(const v8h*)(pb0 + 16);
        v8h b1l = *(const v8h*)(pb1);
        v8h b1h = *(const v8h*)(pb1 + 16);
        pa0 += 32; pa1 += 32; pb0 += 32; pb1 += 32;
        __builtin_prefetch(pa0, 0, 3);   // global_prefetch_b8: next k-slabs
        __builtin_prefetch(pa1, 0, 3);
        __builtin_prefetch(pb0, 0, 3);
        __builtin_prefetch(pb1, 0, 3);

        v16h a0 = __builtin_shufflevector(a0l, a0h, 0,1,2,3,4,5,6,7,8,9,10,11,12,13,14,15);
        v16h a1 = __builtin_shufflevector(a1l, a1h, 0,1,2,3,4,5,6,7,8,9,10,11,12,13,14,15);
        v16h b0 = __builtin_shufflevector(b0l, b0h, 0,1,2,3,4,5,6,7,8,9,10,11,12,13,14,15);
        v16h b1 = __builtin_shufflevector(b1l, b1h, 0,1,2,3,4,5,6,7,8,9,10,11,12,13,14,15);

        acc00 = __builtin_amdgcn_wmma_f32_16x16x32_f16(false, a0, false, b0,
                                                       (short)0, acc00, false, false);
        acc01 = __builtin_amdgcn_wmma_f32_16x16x32_f16(false, a0, false, b1,
                                                       (short)0, acc01, false, false);
        acc10 = __builtin_amdgcn_wmma_f32_16x16x32_f16(false, a1, false, b0,
                                                       (short)0, acc10, false, false);
        acc11 = __builtin_amdgcn_wmma_f32_16x16x32_f16(false, a1, false, b1,
                                                       (short)0, acc11, false, false);
    }

#pragma unroll
    for (int j = 0; j < 8; ++j) {
        const int r0 = m0 + hl * 8 + j;
        const int r1 = r0 + 16;
        size_t i00 = (size_t)r0 * N + n0 + lr;
        size_t i01 = i00 + 16;
        size_t i10 = (size_t)r1 * N + n0 + lr;
        size_t i11 = i10 + 16;
        float v00 = acc00[j], v01 = acc01[j], v10 = acc10[j], v11 = acc11[j];
        if (resid) {
            v00 += resid[i00]; v01 += resid[i01];
            v10 += resid[i10]; v11 += resid[i11];
        }
        C[i00] = v00; C[i01] = v01; C[i10] = v10; C[i11] = v11;
    }
}

// ---------------------------------------------------------------------------
// 4. Depthwise causal conv (D_CONV=4) + bias + SiLU.
// ---------------------------------------------------------------------------
__global__ __launch_bounds__(256) void conv_silu_k(
    const float* __restrict__ xz, const float* __restrict__ cw,
    const float* __restrict__ cb, float* __restrict__ xbc)
{
    size_t idx = (size_t)blockIdx.x * blockDim.x + threadIdx.x;
    if (idx >= (size_t)NTOK * D_INNER) return;
    const int d   = (int)(idx & (D_INNER - 1));
    const int tok = (int)(idx >> 11);
    const int t   = tok & (SEQ - 1);
    const int bb  = tok >> 11;

    float acc = cb[d];
    const float* w = cw + d * D_CONV;
#pragma unroll
    for (int j = 0; j < D_CONV; ++j) {
        int tt = t - (D_CONV - 1) + j;
        if (tt >= 0)
            acc += w[j] * xz[(size_t)(bb * SEQ + tt) * (2 * D_INNER) + d];
    }
    float sg = 1.f / (1.f + __expf(-acc));
    xbc[idx] = acc * sg;
}

// ---------------------------------------------------------------------------
// 5. x_proj: params[tok][j] = sum_d xbc[tok][d] * W[d][j]   (N=33, VALU dots)
// ---------------------------------------------------------------------------
__global__ __launch_bounds__(64) void xproj_k(
    const float* __restrict__ xbc, const float* __restrict__ w,
    float* __restrict__ params)
{
    const int tok = blockIdx.x;
    const int j   = threadIdx.x;
    if (j >= NPROJ) return;
    const float* row = xbc + (size_t)tok * D_INNER;
    float acc = 0.f;
    for (int d = 0; d < D_INNER; ++d)
        acc += row[d] * w[d * NPROJ + j];
    params[(size_t)tok * NPROJ + j] = acc;
}

// ---------------------------------------------------------------------------
// Chunked linear-recurrence scan.  h_t = exp(dt_t*A_s)*h_{t-1} + dt_t*u_t*B_t[s]
// is linear in h, and since A_s is constant the chunk decay is exp(A_s * sum dt).
// Pass 1: per (b,d,chunk,state) local scan with h0=0 -> chunk-final h + decay.
// Pass 2: per (b,d,state) serial 16-chunk carry combine (tiny).
// Pass 3: per (b,d,chunk) re-scan from the true incoming state, produce y via
//         width-16 __shfl_xor reduction across the 16 states.
// tid bit layout for passes 1/3: [3:0]=s  [7:4]=chunk  [18:8]=d  [19]=batch.
// ---------------------------------------------------------------------------
__global__ __launch_bounds__(256) void scan_pass1_k(
    const float* __restrict__ xbc, const float* __restrict__ params,
    const float* __restrict__ A_log, const float* __restrict__ dt_w,
    const float* __restrict__ dt_b,
    float* __restrict__ hfin, float* __restrict__ decay)
{
    const int tid = blockIdx.x * 256 + threadIdx.x;
    if (tid >= BATCH * D_INNER * NCHUNK * D_STATE) return;
    const int s  = tid & (D_STATE - 1);
    const int c  = (tid >> 4) & (NCHUNK - 1);
    const int d  = (tid >> 8) & (D_INNER - 1);
    const int bb = tid >> 19;

    const float As = -__expf(A_log[d * D_STATE + s]);
    const float dw = dt_w[d];
    const float db = dt_b[d];

    float h = 0.f, dtsum = 0.f;
    const size_t tok0 = (size_t)bb * SEQ + c * CHUNK;
    for (int t = 0; t < CHUNK; ++t) {
        const size_t tok = tok0 + t;
        const float* pr = params + tok * NPROJ;
        float xv = pr[2 * D_STATE] * dw + db;
        float dt = (xv > 20.f) ? xv : log1pf(__expf(xv));
        float u  = xbc[tok * D_INNER + d];
        h = __expf(dt * As) * h + dt * u * pr[s];
        dtsum += dt;
    }
    hfin[tid]  = h;
    decay[tid] = __expf(As * dtsum);
}

__global__ __launch_bounds__(256) void scan_pass2_k(
    const float* __restrict__ hfin, const float* __restrict__ decay,
    float* __restrict__ hin)
{
    const int tid = blockIdx.x * 256 + threadIdx.x;
    if (tid >= BATCH * D_INNER * D_STATE) return;
    const int s  = tid & (D_STATE - 1);
    const int d  = (tid >> 4) & (D_INNER - 1);
    const int bb = tid >> 15;

    const size_t base = (((size_t)bb * D_INNER + d) * NCHUNK) * D_STATE + s;
    float carry = 0.f;
#pragma unroll
    for (int c = 0; c < NCHUNK; ++c) {
        const size_t idx = base + (size_t)c * D_STATE;
        hin[idx] = carry;
        carry = decay[idx] * carry + hfin[idx];
    }
}

__global__ __launch_bounds__(256) void scan_pass3_k(
    const float* __restrict__ xbc, const float* __restrict__ params,
    const float* __restrict__ A_log, const float* __restrict__ dt_w,
    const float* __restrict__ dt_b, const float* __restrict__ hin,
    float* __restrict__ xz)
{
    const int tid = blockIdx.x * 256 + threadIdx.x;
    if (tid >= BATCH * D_INNER * NCHUNK * D_STATE) return;
    const int s  = tid & (D_STATE - 1);
    const int c  = (tid >> 4) & (NCHUNK - 1);
    const int d  = (tid >> 8) & (D_INNER - 1);
    const int bb = tid >> 19;

    const float As = -__expf(A_log[d * D_STATE + s]);
    const float dw = dt_w[d];
    const float db = dt_b[d];

    float h = hin[tid];   // true incoming state for this chunk
    const size_t tok0 = (size_t)bb * SEQ + c * CHUNK;
    for (int t = 0; t < CHUNK; ++t) {
        const size_t tok = tok0 + t;
        const float* pr = params + tok * NPROJ;
        float xv = pr[2 * D_STATE] * dw + db;
        float dt = (xv > 20.f) ? xv : log1pf(__expf(xv));
        float u  = xbc[tok * D_INNER + d];
        h = __expf(dt * As) * h + dt * u * pr[s];
        float yc = h * pr[D_STATE + s];
        // sum over the 16 states held by this 16-lane group
        yc += __shfl_xor(yc, 8, 16);
        yc += __shfl_xor(yc, 4, 16);
        yc += __shfl_xor(yc, 2, 16);
        yc += __shfl_xor(yc, 1, 16);
        if (s == 0)
            xz[tok * (2 * D_INNER) + d] = yc;   // y over dead x_b half of xz
    }
}

// ---------------------------------------------------------------------------
// 7. Gate y * silu(z) and convert to f16 for the output GEMM.
// ---------------------------------------------------------------------------
__global__ __launch_bounds__(256) void gate_k(
    const float* __restrict__ xz, _Float16* __restrict__ yg)
{
    size_t idx = (size_t)blockIdx.x * blockDim.x + threadIdx.x;
    if (idx >= (size_t)NTOK * D_INNER) return;
    const int d      = (int)(idx & (D_INNER - 1));
    const size_t tok = idx >> 11;
    float y = xz[tok * (2 * D_INNER) + d];
    float z = xz[tok * (2 * D_INNER) + D_INNER + d];
    float sg = z / (1.f + __expf(-z));
    yg[idx] = (_Float16)(y * sg);
}

// ---------------------------------------------------------------------------
// Launch
// ---------------------------------------------------------------------------
extern "C" void kernel_launch(void* const* d_in, const int* in_sizes, int n_in,
                              void* d_out, int out_size, void* d_ws, size_t ws_size,
                              hipStream_t stream)
{
    (void)in_sizes; (void)n_in; (void)out_size; (void)ws_size;
    const float* x        = (const float*)d_in[0];
    const float* in_proj  = (const float*)d_in[1];
    const float* conv_w   = (const float*)d_in[2];
    const float* conv_b   = (const float*)d_in[3];
    const float* x_proj   = (const float*)d_in[4];
    const float* A_log    = (const float*)d_in[5];
    const float* dt_w     = (const float*)d_in[6];
    const float* dt_b     = (const float*)d_in[7];
    const float* out_proj = (const float*)d_in[8];
    const float* ln_g     = (const float*)d_in[9];
    const float* ln_b     = (const float*)d_in[10];
    float* out = (float*)d_out;

    char* ws = (char*)d_ws;
    size_t off = 0;
    auto take = [&](size_t bytes) -> char* {
        char* p = ws + off;
        off = (off + bytes + 255) & ~(size_t)255;
        return p;
    };
    _Float16* xn   = (_Float16*)take((size_t)NTOK * D_MODEL * sizeof(_Float16));
    _Float16* wT1  = (_Float16*)take((size_t)(2 * D_INNER) * D_MODEL * sizeof(_Float16));
    _Float16* wT2  = (_Float16*)take((size_t)D_MODEL * D_INNER * sizeof(_Float16));
    float*    xz   = (float*)take((size_t)NTOK * (2 * D_INNER) * sizeof(float));
    float*    xbc  = (float*)take((size_t)NTOK * D_INNER * sizeof(float));
    float*    par  = (float*)take((size_t)NTOK * NPROJ * sizeof(float));
    _Float16* yg   = (_Float16*)take((size_t)NTOK * D_INNER * sizeof(_Float16));
    const size_t nscan = (size_t)BATCH * D_INNER * NCHUNK * D_STATE; // 1M
    float*    hfin  = (float*)take(nscan * sizeof(float));
    float*    decay = (float*)take(nscan * sizeof(float));
    float*    hin   = (float*)take(nscan * sizeof(float));

    // 1. LayerNorm -> f16
    ln_to_f16_k<<<NTOK, 256, 0, stream>>>(x, ln_g, ln_b, xn);

    // 2. Weight transposes (f32 KxN -> f16 NxK)
    transpose_to_f16_k<<<(int)(((size_t)D_MODEL * 2 * D_INNER + 255) / 256), 256, 0, stream>>>(
        in_proj, wT1, D_MODEL, 2 * D_INNER);
    transpose_to_f16_k<<<(int)(((size_t)D_INNER * D_MODEL + 255) / 256), 256, 0, stream>>>(
        out_proj, wT2, D_INNER, D_MODEL);

    // 3. in_proj GEMM: xz = xn @ in_proj   (4096 x 4096, K=1024)
    wmma_gemm_f16f32_k<<<dim3((2 * D_INNER) / 64, NTOK / 128), 256, 0, stream>>>(
        xn, wT1, xz, nullptr, NTOK, 2 * D_INNER, D_MODEL);

    // 4. Depthwise conv + SiLU
    conv_silu_k<<<(int)(((size_t)NTOK * D_INNER + 255) / 256), 256, 0, stream>>>(
        xz, conv_w, conv_b, xbc);

    // 5. x_proj (B, C, dt params)
    xproj_k<<<NTOK, 64, 0, stream>>>(xbc, x_proj, par);

    // 6. Chunked selective scan (3 passes; y overwrites x_b half of xz)
    scan_pass1_k<<<(int)(nscan / 256), 256, 0, stream>>>(
        xbc, par, A_log, dt_w, dt_b, hfin, decay);
    scan_pass2_k<<<(BATCH * D_INNER * D_STATE) / 256, 256, 0, stream>>>(
        hfin, decay, hin);
    scan_pass3_k<<<(int)(nscan / 256), 256, 0, stream>>>(
        xbc, par, A_log, dt_w, dt_b, hin, xz);

    // 7. Gate + f16 convert
    gate_k<<<(int)(((size_t)NTOK * D_INNER + 255) / 256), 256, 0, stream>>>(xz, yg);

    // 8. out_proj GEMM + residual: out = yg @ out_proj + x
    wmma_gemm_f16f32_k<<<dim3(D_MODEL / 64, NTOK / 128), 256, 0, stream>>>(
        yg, wT2, out, x, NTOK, D_MODEL, D_INNER);
}
